// GFootballMamba_72026601554654
// MI455X (gfx1250) — matrix operand order repaired and checked by reading, not compiled, not run on hardware
//
#include <hip/hip_runtime.h>
#include <math.h>

typedef __attribute__((ext_vector_type(16))) __bf16 v16bf;
typedef __attribute__((ext_vector_type(8)))  float  v8f;

#define BB   64
#define TT   512
#define TOKN (BB*TT)

// ---------------- WMMA helpers (A/B in contiguous k order, k = 16*half + s; D per CDNA5 ISA 7.12.2) ----------------

__device__ inline v8f wmma_bf(v16bf a, v16bf b, v8f c) {
  return __builtin_amdgcn_wmma_f32_16x16x32_bf16(false, a, false, b, (short)0, c,
                                                 false, false);
}

// Stage A: 16 x (Kfr*32) f32 tile (LDS, leading dim ld) -> fragment-native bf16
// LDS buffer. One conversion pass, reused by every n-tile (and across GEMMs).
// lane 0-15: M=lane, K in {half*16+0..15}, half = lane>>4.
__device__ inline void stageA(const float* __restrict__ fbuf, int ld, int Kfr,
                              __bf16* __restrict__ abf, int lane) {
  int m = lane & 15, half = lane >> 4;
  for (int kf = 0; kf < Kfr; ++kf) {
    v16bf a;
#pragma unroll
    for (int i = 0; i < 8; ++i) {
      int kb = kf * 32 + half * 16 + i * 2;
      float2 f = *(const float2*)&fbuf[m * ld + kb];   // kb even, 8B aligned
      a[2 * i]     = (__bf16)f.x;
      a[2 * i + 1] = (__bf16)f.y;
    }
    *(v16bf*)(abf + ((size_t)kf * 32 + lane) * 16) = a;
  }
}

// D fragment (16x16 f32) -> LDS buffer. VGPR r: M=r (lanes 0-15) / r+8 (16-31).
__device__ inline void stD(float* buf, int ld, int n0, v8f d, int lane) {
  int n = n0 + (lane & 15), mh = (lane >> 4) * 8;
#pragma unroll
  for (int r = 0; r < 8; ++r) buf[(mh + r) * ld + n] = d[r];
}

// One-wave GEMM: bf16-staged A (LDS) x pre-packed bf16 weights (global).
// Inner loop: 2x ds_load_b128 + 2x global_load_b128 + 1x WMMA. No cvt, no
// bounds checks, no divergence.
__device__ inline void gemm_staged(const __bf16* __restrict__ abf, int Kfr,
                                   const __bf16* __restrict__ Wp, int Ntiles,
                                   const float* bias, int nmax,
                                   float* obuf, int old, int lane) {
  for (int nt = 0; nt < Ntiles; ++nt) {
    v8f acc = {};
    for (int kf = 0; kf < Kfr; ++kf) {
      v16bf a = *(const v16bf*)(abf + ((size_t)kf * 32 + lane) * 16);
      v16bf b = *(const v16bf*)(Wp + ((size_t)(nt * Kfr + kf) * 32 + lane) * 16);
      acc = wmma_bf(a, b, acc);
    }
    if (bias) {
      int n = nt * 16 + (lane & 15);
      float bv = (n < nmax) ? bias[n] : 0.f;
#pragma unroll
      for (int r = 0; r < 8; ++r) acc[r] += bv;
    }
    stD(obuf, old, nt * 16, acc, lane);
  }
}

__device__ inline float wred(float v) {
#pragma unroll
  for (int m = 16; m > 0; m >>= 1) v += __shfl_xor(v, m, 32);
  return v;
}

// LayerNorm over rows of buf (16 tokens x F), optional SiLU epilogue.
__device__ inline void ln_rows(float* buf, int ld, int F, const float* g,
                               const float* bta, int lane, bool silu) {
  for (int m = 0; m < 16; ++m) {
    float s = 0.f, s2 = 0.f;
    for (int j = lane; j < F; j += 32) {
      float v = buf[m * ld + j];
      s += v; s2 += v * v;
    }
    s = wred(s); s2 = wred(s2);
    float mean = s / (float)F;
    float var  = s2 / (float)F - mean * mean;
    float inv  = rsqrtf(var + 1e-5f);
    for (int j = lane; j < F; j += 32) {
      float v = (buf[m * ld + j] - mean) * inv * g[j] + bta[j];
      if (silu) v = v / (1.f + __expf(-v));
      buf[m * ld + j] = v;
    }
  }
}

// ---------------- Weight packing: f32 row-major -> fragment-native bf16 ------

__global__ void __launch_bounds__(32)
k_pack(const float* __restrict__ W, int ldk, int kmax, int nmax, int Kfr,
       __bf16* __restrict__ dst) {
  int lane = threadIdx.x & 31;
  int frag = blockIdx.x;                 // nt*Kfr + kf
  int nt = frag / Kfr, kf = frag - nt * Kfr;
  int n = nt * 16 + (lane & 15), half = lane >> 4;
  v16bf b;
#pragma unroll
  for (int i = 0; i < 8; ++i) {
    int k = kf * 32 + half * 16 + 2 * i;
    float f0 = (n < nmax && (k)     < kmax) ? W[(size_t)n * ldk + k]     : 0.f;
    float f1 = (n < nmax && (k + 1) < kmax) ? W[(size_t)n * ldk + k + 1] : 0.f;
    b[2 * i]     = (__bf16)f0;
    b[2 * i + 1] = (__bf16)f1;
  }
  *(v16bf*)(dst + ((size_t)frag * 32 + lane) * 16) = b;
}

// Pack w_b (rows 0..7) and w_c (rows 8..15) into one N=16 tile (Kfr=4).
__global__ void __launch_bounds__(32)
k_pack_bc(const float* __restrict__ wb, const float* __restrict__ wc,
          __bf16* __restrict__ dst) {
  int lane = threadIdx.x & 31;
  int kf = blockIdx.x;                   // 0..3
  int n = lane & 15, half = lane >> 4;
  const float* W = (n < 8) ? wb : wc;
  int nr = n & 7;
  v16bf b;
#pragma unroll
  for (int i = 0; i < 8; ++i) {
    int k = kf * 32 + half * 16 + 2 * i;
    b[2 * i]     = (__bf16)W[nr * 128 + k];
    b[2 * i + 1] = (__bf16)W[nr * 128 + k + 1];
  }
  *(v16bf*)(dst + ((size_t)kf * 32 + lane) * 16) = b;
}

// ---------------- Kernel 1: encoder + action embedding + projection ----------

__global__ void __launch_bounds__(32)
k_encode(const float* __restrict__ obs, const int* __restrict__ acts,
         const float* __restrict__ emb,
         const __bf16* pew1, const float* eb1, const float* eg1, const float* ebb1,
         const __bf16* pew2, const float* eb2, const float* eg2, const float* ebb2,
         const __bf16* ppw,  const float* pb,  const float* pg,  const float* pbb,
         float* __restrict__ xout) {
  __shared__ float bufA[16 * 256];
  __shared__ float bufB[16 * 256];
  __shared__ __bf16 abuf[8 * 32 * 16];
  int lane = threadIdx.x & 31;
  int m0 = blockIdx.x * 16;

  // stage obs rows, K padded 115 -> 128 with zeros
  for (int idx = lane; idx < 16 * 128; idx += 32) {
    int m = idx >> 7, k = idx & 127;
    bufA[m * 256 + k] = (k < 115) ? obs[(size_t)(m0 + m) * 115 + k] : 0.f;
  }
  stageA(bufA, 256, 4, abuf, lane);
  gemm_staged(abuf, 4, pew1, 16, eb1, 256, bufB, 256, lane);
  ln_rows(bufB, 256, 256, eg1, ebb1, lane, true);          // SiLU(LN(enc1))
  stageA(bufB, 256, 8, abuf, lane);
  gemm_staged(abuf, 8, pew2, 8, eb2, 128, bufA, 256, lane);
  ln_rows(bufA, 256, 128, eg2, ebb2, lane, false);         // LN(enc2)

  // concat action embedding (cols 128..143), zero pad (144..159)
  for (int idx = lane; idx < 16 * 32; idx += 32) {
    int m = idx >> 5, j = idx & 31;
    float v = 0.f;
    if (j < 16) v = emb[(size_t)acts[m0 + m] * 16 + j];
    bufA[m * 256 + 128 + j] = v;
  }
  stageA(bufA, 256, 5, abuf, lane);
  gemm_staged(abuf, 5, ppw, 8, pb, 128, bufB, 256, lane);
  ln_rows(bufB, 256, 128, pg, pbb, lane, true);            // SiLU(LN(proj))

  for (int idx = lane; idx < 16 * 128; idx += 32) {
    int m = idx >> 7, d = idx & 127;
    xout[(size_t)(m0 + m) * 128 + d] = bufB[m * 256 + d];
  }
}

// ---------------- Kernel 2: per-layer pre-scan (LN, w_ig, dt, b, c) ----------

__global__ void __launch_bounds__(32)
k_layer_pre(const float* __restrict__ x,
            const float* lng, const float* lnb,
            const __bf16* pwig, const __bf16* pwdt, const float* bdt,
            const __bf16* pbc,
            float* __restrict__ x_in, float* __restrict__ sz,
            float* __restrict__ dtx,  float* __restrict__ bc) {
  __shared__ float bufA[16 * 256];
  __shared__ float bufB[16 * 256];
  __shared__ __bf16 abuf[8 * 32 * 16];
  int lane = threadIdx.x & 31;
  int m0 = blockIdx.x * 16;

  for (int idx = lane; idx < 16 * 128; idx += 32) {
    int m = idx >> 7, d = idx & 127;
    bufA[m * 256 + d] = x[(size_t)(m0 + m) * 128 + d];
  }
  ln_rows(bufA, 256, 128, lng, lnb, lane, false);
  stageA(bufA, 256, 4, abuf, lane);
  // xp = xn @ w_ig^T : N=256
  gemm_staged(abuf, 4, pwig, 16, nullptr, 256, bufB, 256, lane);

  // split: x_in = xp[:, :128]; sz = silu(xp[:, 128:])
  for (int idx = lane; idx < 16 * 128; idx += 32) {
    int m = idx >> 7, d = idx & 127;
    size_t n = (size_t)(m0 + m) * 128 + d;
    float xi = bufB[m * 256 + d];
    float zv = bufB[m * 256 + 128 + d];
    x_in[n] = xi;
    sz[n] = zv / (1.f + __expf(-zv));
  }

  // stage x_in once; reused by both the w_dt and [b|c] GEMMs
  stageA(bufB, 256, 4, abuf, lane);

  // dt = softplus(x_in @ w_dt^T + b_dt); interleave {dt, x_in} for the scan
  gemm_staged(abuf, 4, pwdt, 8, bdt, 128, bufA, 256, lane);
  for (int idx = lane; idx < 16 * 128; idx += 32) {
    int m = idx >> 7, d = idx & 127;
    float v = bufA[m * 256 + d];
    float sp = (v > 20.f) ? v : log1pf(__expf(v));
    size_t n = (size_t)(m0 + m) * 128 + d;
    dtx[n * 2]     = sp;
    dtx[n * 2 + 1] = bufB[m * 256 + d];
  }

  // [bt | ct] = x_in @ [w_b; w_c]^T in one N=16 tile; interleave {b, c}
  gemm_staged(abuf, 4, pbc, 1, nullptr, 16, bufA, 256, lane);
  for (int idx = lane; idx < 16 * 8; idx += 32) {
    int m = idx >> 3, s = idx & 7;
    size_t n = (size_t)(m0 + m) * 8 + s;
    bc[n * 2]     = bufA[m * 256 + s];
    bc[n * 2 + 1] = bufA[m * 256 + 8 + s];
  }
}

// ---------------- Kernel 3: selective-state scan -----------------------------
// lane per (b,d,s); exp(dt*A) in-register; float2 streams (2 loads/step);
// 8-lane shuffle reduction produces y[b,t,d].

__global__ void __launch_bounds__(256)
k_scan(const float* __restrict__ dtx, const float* __restrict__ bc,
       const float* __restrict__ h0,  const float* __restrict__ alog,
       float* __restrict__ y) {
  int g = blockIdx.x * 256 + threadIdx.x;   // 0 .. 65535
  int b = g >> 10;
  int r = g & 1023;
  int d = r >> 3;
  int s = r & 7;
  float h = h0[(size_t)b * 1024 + r];
  float A = -__expf(alog[d * 8 + s]);
  const float2* pdx = (const float2*)dtx + (size_t)b * TT * 128 + d;
  const float2* pbc = (const float2*)bc  + (size_t)b * TT * 8 + s;
  float* py = y + ((size_t)b * TT) * 128 + d;
  for (int t = 0; t < TT; ++t) {
    float2 dx  = *pdx;                     // {dt, x_in}
    float2 bcv = *pbc;                     // {b, c}
    float da = __expf(dx.x * A);
    h = fmaf(da, h, dx.x * dx.y * bcv.x);
    float p = h * bcv.y;
    p += __shfl_xor(p, 1, 32);
    p += __shfl_xor(p, 2, 32);
    p += __shfl_xor(p, 4, 32);
    if (s == 0) *py = p;
    pdx += 128; pbc += 8; py += 128;
  }
}

// ---------------- Kernel 4: per-layer post-scan (gate, w_out, residual) ------

__global__ void __launch_bounds__(32)
k_layer_post(const float* __restrict__ y,  const float* __restrict__ sz,
             const float* __restrict__ x_in, const float* dpar,
             const __bf16* pwout, float* __restrict__ x) {
  __shared__ float bufA[16 * 256];
  __shared__ float bufB[16 * 256];
  __shared__ __bf16 abuf[8 * 32 * 16];
  int lane = threadIdx.x & 31;
  int m0 = blockIdx.x * 16;

  for (int idx = lane; idx < 16 * 128; idx += 32) {
    int m = idx >> 7, d = idx & 127;
    size_t n = (size_t)(m0 + m) * 128 + d;
    bufA[m * 256 + d] = y[n] * sz[n] + x_in[n] * dpar[d];   // ssm
  }
  stageA(bufA, 256, 4, abuf, lane);
  gemm_staged(abuf, 4, pwout, 8, nullptr, 128, bufB, 256, lane);
  for (int idx = lane; idx < 16 * 128; idx += 32) {
    int m = idx >> 7, d = idx & 127;
    x[(size_t)(m0 + m) * 128 + d] += bufB[m * 256 + d];     // residual
  }
}

// ---------------- Kernel 5: final LN + policy head ---------------------------

__global__ void __launch_bounds__(32)
k_head(const float* __restrict__ x, const float* fng, const float* fnb,
       const __bf16* ppw1, const float* pb1, const float* pg, const float* pbb,
       const __bf16* ppw2, const float* pb2, float* __restrict__ out) {
  __shared__ float bufA[16 * 256];
  __shared__ float bufB[16 * 256];
  __shared__ __bf16 abuf[8 * 32 * 16];
  int lane = threadIdx.x & 31;
  int m0 = blockIdx.x * 16;

  for (int idx = lane; idx < 16 * 128; idx += 32) {
    int m = idx >> 7, d = idx & 127;
    bufA[m * 256 + d] = x[(size_t)(m0 + m) * 128 + d];
  }
  ln_rows(bufA, 256, 128, fng, fnb, lane, false);
  stageA(bufA, 256, 4, abuf, lane);
  gemm_staged(abuf, 4, ppw1, 8, pb1, 128, bufB, 256, lane);
  ln_rows(bufB, 256, 128, pg, pbb, lane, true);             // SiLU(LN)
  stageA(bufB, 256, 4, abuf, lane);
  gemm_staged(abuf, 4, ppw2, 2, pb2, 19, bufA, 256, lane);
  for (int idx = lane; idx < 16 * 32; idx += 32) {
    int m = idx >> 5, j = idx & 31;
    if (j < 19) out[(size_t)(m0 + m) * 19 + j] = bufA[m * 256 + j];
  }
}

// ---------------- host launch ------------------------------------------------

extern "C" void kernel_launch(void* const* d_in, const int* in_sizes, int n_in,
                              void* d_out, int out_size, void* d_ws,
                              size_t ws_size, hipStream_t stream) {
  const float* obs   = (const float*)d_in[0];
  const float* h0    = (const float*)d_in[1];
  const float* ew1   = (const float*)d_in[2];
  const float* eb1   = (const float*)d_in[3];
  const float* eg1   = (const float*)d_in[4];
  const float* ebb1  = (const float*)d_in[5];
  const float* ew2   = (const float*)d_in[6];
  const float* eb2   = (const float*)d_in[7];
  const float* eg2   = (const float*)d_in[8];
  const float* ebb2  = (const float*)d_in[9];
  const float* emb   = (const float*)d_in[10];
  const float* pw    = (const float*)d_in[11];
  const float* pb    = (const float*)d_in[12];
  const float* pg    = (const float*)d_in[13];
  const float* pbb   = (const float*)d_in[14];
  const float* mlng  = (const float*)d_in[15];
  const float* mlnb  = (const float*)d_in[16];
  const float* mwig  = (const float*)d_in[17];
  const float* mwdt  = (const float*)d_in[18];
  const float* mbdt  = (const float*)d_in[19];
  const float* malog = (const float*)d_in[20];
  const float* mwb   = (const float*)d_in[21];
  const float* mwc   = (const float*)d_in[22];
  const float* md    = (const float*)d_in[23];
  const float* mwout = (const float*)d_in[24];
  const float* fng   = (const float*)d_in[25];
  const float* fnb   = (const float*)d_in[26];
  const float* pw1   = (const float*)d_in[27];
  const float* pb1   = (const float*)d_in[28];
  const float* polg  = (const float*)d_in[29];
  const float* polbb = (const float*)d_in[30];
  const float* pw2   = (const float*)d_in[31];
  const float* pb2   = (const float*)d_in[32];
  const int*   acts  = (const int*)d_in[33];
  float* out = (float*)d_out;

  // ws layout: [packed bf16 weights | f32 activation buffers]
  __bf16* P = (__bf16*)d_ws;
  const size_t FR = 512;                  // elements per fragment (32 lanes x 16)
  __bf16* Pew1  = P;                      // 64 frags
  __bf16* Pew2  = Pew1 + 64 * FR;         // 64
  __bf16* Pproj = Pew2 + 64 * FR;         // 40
  __bf16* Plyr  = Pproj + 40 * FR;        // 3 * 132
  __bf16* Ppol1 = Plyr + 3 * 132 * FR;    // 32
  __bf16* Ppol2 = Ppol1 + 32 * FR;        // 8  -> total 604 frags (~0.6 MB)

  float* F   = (float*)d_ws + 160 * 1024; // past packed region, aligned
  float* X   = F;                          // TOKN*128
  float* XIN = X   + (size_t)TOKN * 128;
  float* SZ  = XIN + (size_t)TOKN * 128;
  float* DTX = SZ  + (size_t)TOKN * 128;   // TOKN*256 interleaved {dt,x}
  float* Y   = DTX + (size_t)TOKN * 256;
  float* BC  = Y   + (size_t)TOKN * 128;   // TOKN*16 interleaved {b,c}

  dim3 bWave(32);
  // ---- pack weights (fragment-native bf16, padded) ----
  k_pack<<<dim3(64), bWave, 0, stream>>>(ew1, 115, 115, 256, 4, Pew1);
  k_pack<<<dim3(64), bWave, 0, stream>>>(ew2, 256, 256, 128, 8, Pew2);
  k_pack<<<dim3(40), bWave, 0, stream>>>(pw, 144, 144, 128, 5, Pproj);
  for (int l = 0; l < 3; ++l) {
    __bf16* L = Plyr + (size_t)l * 132 * FR;
    k_pack<<<dim3(64), bWave, 0, stream>>>(mwig + (size_t)l * 256 * 128,
                                           128, 128, 256, 4, L);            // wig
    k_pack<<<dim3(32), bWave, 0, stream>>>(mwdt + (size_t)l * 128 * 128,
                                           128, 128, 128, 4, L + 64 * FR);  // wdt
    k_pack_bc<<<dim3(4), bWave, 0, stream>>>(mwb + l * 1024, mwc + l * 1024,
                                             L + 96 * FR);                  // b|c
    k_pack<<<dim3(32), bWave, 0, stream>>>(mwout + (size_t)l * 128 * 128,
                                           128, 128, 128, 4, L + 100 * FR); // wout
  }
  k_pack<<<dim3(32), bWave, 0, stream>>>(pw1, 128, 128, 128, 4, Ppol1);
  k_pack<<<dim3(8),  bWave, 0, stream>>>(pw2, 128, 128, 19, 4, Ppol2);

  dim3 gTiles(TOKN / 16);
  k_encode<<<gTiles, bWave, 0, stream>>>(obs, acts, emb,
      Pew1, eb1, eg1, ebb1, Pew2, eb2, eg2, ebb2, Pproj, pb, pg, pbb, X);

  for (int l = 0; l < 3; ++l) {
    __bf16* L = Plyr + (size_t)l * 132 * FR;
    k_layer_pre<<<gTiles, bWave, 0, stream>>>(X,
        mlng + l * 128, mlnb + l * 128,
        L, L + 64 * FR, mbdt + l * 128, L + 96 * FR,
        XIN, SZ, DTX, BC);
    k_scan<<<dim3(256), dim3(256), 0, stream>>>(DTX, BC,
        h0 + (size_t)l * BB * 1024, malog + l * 1024, Y);
    k_layer_post<<<gTiles, bWave, 0, stream>>>(Y, SZ, XIN,
        md + l * 128, L + 100 * FR, X);
  }

  k_head<<<gTiles, bWave, 0, stream>>>(X, fng, fnb,
      Ppol1, pb1, polg, polbb, Ppol2, pb2, out);
}
